// MaskRCNN_51823075394227
// MI455X (gfx1250) — matrix operand order repaired and checked by reading, not compiled, not run on hardware
//
#include <hip/hip_runtime.h>
#include <hip/hip_bf16.h>
#include <math.h>

typedef __attribute__((ext_vector_type(16))) __bf16 bf16x16;
typedef __attribute__((ext_vector_type(8)))  float   f32x8;

#define BN_EPS 1e-5f

// ---------------------------------------------------------------------------
// Layout: NCHW -> NHWC with channel zero-padding (C -> Cp)
// ---------------------------------------------------------------------------
__global__ void nchw_to_nhwc_pad_k(const float* __restrict__ in, float* __restrict__ out,
                                   int N, int C, int Cp, int H, int W) {
  long idx = (long)blockIdx.x * blockDim.x + threadIdx.x;
  long tot = (long)N * H * W * Cp;
  if (idx >= tot) return;
  int c = idx % Cp; long r = idx / Cp;
  int w = r % W; r /= W;
  int h = r % H; int n = (int)(r / H);
  out[idx] = (c < C) ? in[(((long)n * C + c) * H + h) * W + w] : 0.f;
}

// ---------------------------------------------------------------------------
// Weight pack: OIHW fp32 -> WMMA fragment-ordered bf16 tiles, with channel
// zero-padding Ci -> Cp (Cp%32==0). K = KH*KW*Cp. Layout:
//   wp[((chunk*(O/16) + tn)*512) + l16*32 + lh*16 + e]
// where chunk=k/32, lh=(k%32)/16, e=k%16, tn=o/16, l16=o%16.
// A lane's 16-element B fragment is then 32 contiguous bytes.
// ---------------------------------------------------------------------------
__global__ void wpack_k(const float* __restrict__ w, __bf16* __restrict__ wp,
                        int O, int Ci, int Cp, int KH, int KW) {
  long idx = (long)blockIdx.x * blockDim.x + threadIdx.x;
  long tot = (long)KH * KW * Cp * O;
  if (idx >= tot) return;
  int o = idx % O; long k = idx / O;
  int ci = k % Cp; long r = k / Cp;
  int kw = r % KW; int kh = (int)(r / KW);
  float v = (ci < Ci) ? w[(((long)o * Ci + ci) * KH + kh) * KW + kw] : 0.f;
  int chunk = (int)(k >> 5); int lh = ((int)k >> 4) & 1; int e = (int)k & 15;
  int tn = o >> 4; int l16 = o & 15;
  wp[(((long)chunk * (O >> 4) + tn) * 512) + l16 * 32 + lh * 16 + e] = (__bf16)v;
}

// mwt [C][O][2][2] (C=K=128) -> packed tiles, cols = (k*2+l)*O + o (Cout=4*O=512)
__global__ void wpack_deconv_k(const float* __restrict__ w, __bf16* __restrict__ wp,
                               int C, int O) {
  long idx = (long)blockIdx.x * blockDim.x + threadIdx.x;
  long tot = (long)C * O * 4;
  if (idx >= tot) return;
  int col = idx % (4 * O); int c = (int)(idx / (4 * O));
  int kl = col / O; int o = col % O;
  float v = w[(((long)c * O + o) * 2 + (kl >> 1)) * 2 + (kl & 1)];
  int chunk = c >> 5; int lh = (c >> 4) & 1; int e = c & 15;
  int tn = col >> 4; int l16 = col & 15;
  wp[(((long)chunk * ((4 * O) >> 4) + tn) * 512) + l16 * 32 + lh * 16 + e] = (__bf16)v;
}

// ---------------------------------------------------------------------------
// Implicit-GEMM conv via WMMA bf16 (f32 acc). Requires Cin%32==0.
// One wave computes a 16 x (16*NT) output tile. No div/mod in the hot loop:
// (kh,kw) are loop induction vars; A-fragment = one contiguous 16-float run;
// B-fragment = one 32-byte contiguous load from the packed weights; the NT
// WMMAs per chunk are independent (no D->A/B hazard).
// SM=0: NHWC store. SM=1: transposed-conv scatter (7x7 -> 14x14, Cout=4*Og).
// ---------------------------------------------------------------------------
template <int NT, int SM>
__global__ void wmma_conv_fast_k(const float* __restrict__ in, const __bf16* __restrict__ wp,
                                 float* __restrict__ out,
                                 int N, int H, int W, int Cin, int Ho, int Wo, int Cout,
                                 int KH, int KW, int stride, int pad) {
  int lane   = threadIdx.x;          // 0..31 (wave32)
  int tileM  = blockIdx.x;
  int tileN0 = blockIdx.y * NT;
  int l16    = lane & 15;
  int lh     = lane >> 4;

  int hw   = Ho * Wo;
  int mrow = tileM * 16 + l16;
  int n    = mrow / hw;
  int rem  = mrow % hw;
  int ho   = rem / Wo;
  int wo   = rem % Wo;
  int hbase = ho * stride - pad;
  int wbase = wo * stride - pad;

  int kcA   = lh ? 16 : 0;           // A-fragment lane K base within chunk
  int CoutT = Cout >> 4;
  const __bf16* wlane = wp + l16 * 32 + lh * 16;

  f32x8 acc[NT];
#pragma unroll
  for (int t = 0; t < NT; ++t) { f32x8 z = {}; acc[t] = z; }

  int chunk = 0;
  for (int kh = 0; kh < KH; ++kh) {
    int hi = hbase + kh;
    bool hok = (hi >= 0) && (hi < H);
    for (int kw = 0; kw < KW; ++kw) {
      int wi = wbase + kw;
      bool ok = hok && (wi >= 0) && (wi < W);
      const float* rowp = in + (((long)n * H + hi) * W + wi) * Cin + kcA;
      for (int ci0 = 0; ci0 < Cin; ci0 += 32, ++chunk) {
        bf16x16 a = {};
        if (ok) {
          const float4* p = (const float4*)(rowp + ci0);
          float4 f0 = p[0], f1 = p[1], f2 = p[2], f3 = p[3];
          a[0]  = (__bf16)f0.x; a[1]  = (__bf16)f0.y; a[2]  = (__bf16)f0.z; a[3]  = (__bf16)f0.w;
          a[4]  = (__bf16)f1.x; a[5]  = (__bf16)f1.y; a[6]  = (__bf16)f1.z; a[7]  = (__bf16)f1.w;
          a[8]  = (__bf16)f2.x; a[9]  = (__bf16)f2.y; a[10] = (__bf16)f2.z; a[11] = (__bf16)f2.w;
          a[12] = (__bf16)f3.x; a[13] = (__bf16)f3.y; a[14] = (__bf16)f3.z; a[15] = (__bf16)f3.w;
        }
        const __bf16* wchunk = wlane + ((long)chunk * CoutT + tileN0) * 512;
#pragma unroll
        for (int t = 0; t < NT; ++t) {
          bf16x16 b = *(const bf16x16*)(wchunk + (long)t * 512);
          acc[t] = __builtin_amdgcn_wmma_f32_16x16x32_bf16(
              false, a, false, b, (short)0, acc[t], false, false);
        }
      }
    }
  }

#pragma unroll
  for (int t = 0; t < NT; ++t) {
    int col = (tileN0 + t) * 16 + l16;
#pragma unroll
    for (int i = 0; i < 8; ++i) {
      int row = tileM * 16 + i + 8 * lh;
      if (SM == 0) {
        out[(long)row * Cout + col] = acc[t][i];
      } else {
        int Og = Cout >> 2;
        int o  = col % Og; int kl = col / Og;
        int k  = kl >> 1;  int l  = kl & 1;
        int n2 = row / hw; int r2 = row % hw;
        int i2 = r2 / Wo;  int j2 = r2 % Wo;
        int HO = 2 * Ho, WO = 2 * Wo;
        out[(((long)n2 * HO + (2 * i2 + k)) * WO + (2 * j2 + l)) * Og + o] = acc[t][i];
      }
    }
  }
}

// ---------------------------------------------------------------------------
// Per-channel mean/var over M rows of [M][C]; mv[0..C-1]=mean, mv[C..2C-1]=var
// ---------------------------------------------------------------------------
__global__ void stats_k(const float* __restrict__ x, float* __restrict__ mv,
                        long M, int C) {
  int c = blockIdx.x;
  float s = 0.f, s2 = 0.f;
  for (long m = threadIdx.x; m < M; m += blockDim.x) {
    float v = x[m * C + c];
    s += v; s2 += v * v;
  }
  __shared__ float sh[256], sh2[256];
  sh[threadIdx.x] = s; sh2[threadIdx.x] = s2;
  __syncthreads();
  for (int o = 128; o > 0; o >>= 1) {
    if ((int)threadIdx.x < o) {
      sh[threadIdx.x]  += sh[threadIdx.x + o];
      sh2[threadIdx.x] += sh2[threadIdx.x + o];
    }
    __syncthreads();
  }
  if (threadIdx.x == 0) {
    float mean = sh[0] / (float)M;
    float var  = sh2[0] / (float)M - mean * mean;
    mv[c] = mean; mv[C + c] = var;
  }
}

__global__ void bn_apply_k(float* __restrict__ x, const float* __restrict__ mv,
                           const float* __restrict__ g, const float* __restrict__ b,
                           const float* __restrict__ res, int relu, long M, int C) {
  long idx = (long)blockIdx.x * blockDim.x + threadIdx.x;
  long tot = M * (long)C;
  if (idx >= tot) return;
  int c = idx % C;
  float v = (x[idx] - mv[c]) * rsqrtf(mv[C + c] + BN_EPS) * g[c] + b[c];
  if (res)  v += res[idx];
  if (relu) v = fmaxf(v, 0.f);
  x[idx] = v;
}

__global__ void bias_act_k(float* __restrict__ x, const float* __restrict__ bias,
                           int relu, long M, int C) {
  long idx = (long)blockIdx.x * blockDim.x + threadIdx.x;
  long tot = M * (long)C;
  if (idx >= tot) return;
  int c = idx % C;
  float v = x[idx] + bias[c];
  if (relu) v = fmaxf(v, 0.f);
  x[idx] = v;
}

// ---------------------------------------------------------------------------
// 3x3 stride-2 pad-1 maxpool, NHWC
// ---------------------------------------------------------------------------
__global__ void maxpool_k(const float* __restrict__ in, float* __restrict__ out,
                          int N, int H, int W, int C, int Ho, int Wo) {
  long idx = (long)blockIdx.x * blockDim.x + threadIdx.x;
  long tot = (long)N * Ho * Wo * C;
  if (idx >= tot) return;
  int c = idx % C; long r = idx / C;
  int wo = r % Wo; r /= Wo;
  int ho = r % Ho; int n = (int)(r / Ho);
  float m = -3.4e38f;
  for (int ky = 0; ky < 3; ++ky) {
    int hi = ho * 2 - 1 + ky;
    if (hi < 0 || hi >= H) continue;
    for (int kx = 0; kx < 3; ++kx) {
      int wi = wo * 2 - 1 + kx;
      if (wi < 0 || wi >= W) continue;
      float v = in[(((long)n * H + hi) * W + wi) * C + c];
      m = fmaxf(m, v);
    }
  }
  out[idx] = m;
}

// ---------------------------------------------------------------------------
// Tiny 1x1 conv (naive): out[m][o] = b[o] + sum_c in[m][c]*w[o][c]
// ---------------------------------------------------------------------------
__global__ void conv1x1_small_k(const float* __restrict__ in, const float* __restrict__ w,
                                const float* __restrict__ b, float* __restrict__ out,
                                long M, int C, int O) {
  long idx = (long)blockIdx.x * blockDim.x + threadIdx.x;
  long tot = M * (long)O;
  if (idx >= tot) return;
  int o = idx % O; long m = idx / O;
  float s = b[o];
  const float* src = in + m * C;
  const float* wr  = w + (long)o * C;
  for (int c = 0; c < C; ++c) s += src[c] * wr[c];
  out[idx] = s;
}

// ---------------------------------------------------------------------------
// Anchors + box decode + normalized (y,x,y,x)/223 boxes
// ---------------------------------------------------------------------------
__global__ void boxes_k(const float* __restrict__ deltas, float* __restrict__ boxes,
                        float* __restrict__ nb, int B, int A) {
  int idx = blockIdx.x * blockDim.x + threadIdx.x;
  if (idx >= B * A) return;
  int a = idx % A;
  int i = a / 14, j = a % 14;
  float acx = j * 16.f + 8.f, acy = i * 16.f + 8.f;     // base=16, scale=2 -> half=16
  float d0 = deltas[idx * 4 + 0], d1 = deltas[idx * 4 + 1];
  float d2 = deltas[idx * 4 + 2], d3 = deltas[idx * 4 + 3];
  float cx = acx + d0 * 32.f, cy = acy + d1 * 32.f;
  float w  = 32.f * expf(d2), h = 32.f * expf(d3);
  float x1 = cx - 0.5f * w, y1 = cy - 0.5f * h;
  float x2 = cx + 0.5f * w, y2 = cy + 0.5f * h;
  boxes[idx * 4 + 0] = x1; boxes[idx * 4 + 1] = y1;
  boxes[idx * 4 + 2] = x2; boxes[idx * 4 + 3] = y2;
  nb[idx * 4 + 0] = y1 / 223.f; nb[idx * 4 + 1] = x1 / 223.f;
  nb[idx * 4 + 2] = y2 / 223.f; nb[idx * 4 + 3] = x2 / 223.f;
}

// ---------------------------------------------------------------------------
// crop_and_resize: cmap NHWC [B,14,14,C] -> crops NHWC [B*A,7,7,C]
// ---------------------------------------------------------------------------
__global__ void crop_k(const float* __restrict__ cmap, const float* __restrict__ nb,
                       float* __restrict__ crops, int B, int A, int C, int HW, int CR) {
  long idx = (long)blockIdx.x * blockDim.x + threadIdx.x;
  long tot = (long)B * A * CR * CR * C;
  if (idx >= tot) return;
  int c = idx % C; long r = idx / C;
  int ix = r % CR; r /= CR;
  int iy = r % CR; int box = (int)(r / CR);
  int bimg = box / A;
  float y1 = nb[box * 4 + 0], x1 = nb[box * 4 + 1];
  float y2 = nb[box * 4 + 2], x2 = nb[box * 4 + 3];
  float ty = (float)iy / (float)(CR - 1);
  float tx = (float)ix / (float)(CR - 1);
  float ys = (y1 + ty * (y2 - y1)) * (float)(HW - 1);
  float xs = (x1 + tx * (x2 - x1)) * (float)(HW - 1);
  bool vy = (ys >= 0.f) && (ys <= (float)(HW - 1));
  bool vx = (xs >= 0.f) && (xs <= (float)(HW - 1));
  float yc = fminf(fmaxf(ys, 0.f), (float)(HW - 1));
  float xc = fminf(fmaxf(xs, 0.f), (float)(HW - 1));
  int y0 = (int)floorf(yc), x0 = (int)floorf(xc);
  float fy = yc - (float)y0, fx = xc - (float)x0;
  int y1i = min(y0 + 1, HW - 1), x1i = min(x0 + 1, HW - 1);
  const float* f = cmap + (long)bimg * HW * HW * C;
  float v00 = f[((long)y0  * HW + x0 ) * C + c];
  float v01 = f[((long)y0  * HW + x1i) * C + c];
  float v10 = f[((long)y1i * HW + x0 ) * C + c];
  float v11 = f[((long)y1i * HW + x1i) * C + c];
  float v = v00 * (1.f - fy) * (1.f - fx) + v01 * (1.f - fy) * fx +
            v10 * fy * (1.f - fx)         + v11 * fy * fx;
  crops[idx] = (vy && vx) ? v : 0.f;
}

// ---------------------------------------------------------------------------
// Final mask 1x1 conv (128->2) + bias, NHWC -> [box][o][14*14]
// ---------------------------------------------------------------------------
__global__ void mask_final_k(const float* __restrict__ up, const float* __restrict__ w,
                             const float* __restrict__ bias, float* __restrict__ out,
                             int NB, int C, int O) {
  long idx = (long)blockIdx.x * blockDim.x + threadIdx.x;
  long tot = (long)NB * O * 196;
  if (idx >= tot) return;
  int p = idx % 196; long r = idx / 196;
  int o = r % O; int box = (int)(r / O);
  int y = p / 14, x = p % 14;
  const float* src = up + (((long)box * 14 + y) * 14 + x) * C;
  const float* wr  = w + (long)o * C;
  float s = bias[o];
  for (int c = 0; c < C; ++c) s += src[c] * wr[c];
  out[idx] = s;   // ((box*O+o)*196+p)
}

// ---------------------------------------------------------------------------
// Greedy NMS, one block per image. A=196.
// ---------------------------------------------------------------------------
__global__ void nms_k(const float* __restrict__ scores_all, const float* __restrict__ boxes,
                      float* __restrict__ keep_out, int A) {
  int b = blockIdx.x, t = threadIdx.x;
  __shared__ float s[196];
  __shared__ float bx[196 * 4];
  __shared__ int   ord[196];
  __shared__ int   sup[196];
  for (int a = t; a < A; a += blockDim.x) {
    float logit = scores_all[(b * A + a) * 2 + 1];
    s[a] = 1.f / (1.f + expf(-logit));
    for (int q = 0; q < 4; ++q) bx[a * 4 + q] = boxes[(b * A + a) * 4 + q];
    sup[a] = 0;
  }
  __syncthreads();
  for (int a = t; a < A; a += blockDim.x) {
    int rnk = 0; float sa = s[a];
    for (int j = 0; j < A; ++j) {
      float sj = s[j];
      if (sj > sa || (sj == sa && j < a)) rnk++;
    }
    ord[rnk] = a;
  }
  __syncthreads();
  for (int i = 0; i < A; ++i) {
    if (!sup[i]) {
      int ai = ord[i];
      float ax1 = bx[ai * 4], ay1 = bx[ai * 4 + 1], ax2 = bx[ai * 4 + 2], ay2 = bx[ai * 4 + 3];
      float areaA = (ax2 - ax1) * (ay2 - ay1);
      for (int jj = t; jj < A; jj += blockDim.x) {
        if (jj > i) {
          int aj = ord[jj];
          float bx1 = bx[aj * 4], by1 = bx[aj * 4 + 1], bx2 = bx[aj * 4 + 2], by2 = bx[aj * 4 + 3];
          float ix = fmaxf(fminf(ax2, bx2) - fmaxf(ax1, bx1), 0.f);
          float iy = fmaxf(fminf(ay2, by2) - fmaxf(ay1, by1), 0.f);
          float inter = ix * iy;
          float areaB = (bx2 - bx1) * (by2 - by1);
          float iou = inter / (areaA + areaB - inter + 1e-8f);
          if (iou > 0.3f) sup[jj] = 1;
        }
      }
    }
    __syncthreads();
  }
  for (int i = t; i < A; i += blockDim.x) {
    int a = ord[i];
    keep_out[b * A + a] = (!sup[i] && s[a] > 0.5f) ? 1.f : 0.f;
  }
}

// ---------------------------------------------------------------------------
// Host orchestration
// ---------------------------------------------------------------------------
extern "C" void kernel_launch(void* const* d_in, const int* in_sizes, int n_in,
                              void* d_out, int out_size, void* d_ws, size_t ws_size,
                              hipStream_t stream) {
  (void)in_sizes; (void)n_in; (void)out_size; (void)ws_size;
  auto F = [&](int i) { return (const float*)d_in[i]; };

  float* ws = (float*)d_ws;
  size_t off = 0;
  auto alloc = [&](size_t n) { float* p = ws + off; off += (n + 15) & ~(size_t)15; return p; };

  float* XN    = alloc(8L * 224 * 224 * 32);  // NHWC input, C padded 3->32
  float* A0    = alloc(8L * 112 * 112 * 64);  // conv1 out
  float* L0    = alloc(1605632);              // rotating layer buffers (56x56x64 max)
  float* L1    = alloc(1605632);
  float* L2    = alloc(1605632);
  float* L3    = alloc(1605632);
  float* MV    = alloc(1024);                 // mean|var, C<=512
  float* R0    = alloc(8L * 14 * 14 * 32);    // rpn feature
  float* BOX   = alloc(8L * 196 * 4);
  float* NB    = alloc(8L * 196 * 4);
  float* CROPS = alloc(1568L * 49 * 256);
  float* MH1   = alloc(1568L * 49 * 256);
  float* MH2   = alloc(1568L * 49 * 128);
  float* UP    = alloc(1568L * 196 * 128);
  __bf16* WB   = (__bf16*)(ws + off); off += 400000;  // 800k bf16 (max need 589,824)

  float* outF     = (float*)d_out;
  float* o_scores = outF;                 // 8*196*2   = 3136
  float* o_deltas = outF + 3136;          // 8*196*4   = 6272
  float* o_masks  = outF + 9408;          // 8*196*2*196 = 614656
  float* o_keep   = outF + 624064;        // 8*196     = 1568

  // weight prep: packed fragment layout, channel-padded Ci -> Cp (Cp%32==0)
  auto cvtw = [&](const float* w, int O, int Ci, int Cp, int KH, int KW) {
    long tot = (long)KH * KW * Cp * O;
    wpack_k<<<dim3((unsigned)((tot + 255) / 256)), dim3(256), 0, stream>>>(
        w, WB, O, Ci, Cp, KH, KW);
  };
  auto conv = [&](const float* in, float* out, int N, int H, int W, int Ci, int Co,
                  int KH, int KW, int st, int pad, int sm) {
    int Ho = (H + 2 * pad - KH) / st + 1;
    int Wo = (W + 2 * pad - KW) / st + 1;
    int M = N * Ho * Wo;
    if (sm == 1) {
      wmma_conv_fast_k<4, 1><<<dim3((unsigned)(M / 16), (unsigned)((Co >> 4) / 4)),
                               dim3(32), 0, stream>>>(
          in, WB, out, N, H, W, Ci, Ho, Wo, Co, KH, KW, st, pad);
    } else if (Co % 64 == 0) {
      wmma_conv_fast_k<4, 0><<<dim3((unsigned)(M / 16), (unsigned)((Co >> 4) / 4)),
                               dim3(32), 0, stream>>>(
          in, WB, out, N, H, W, Ci, Ho, Wo, Co, KH, KW, st, pad);
    } else {
      wmma_conv_fast_k<2, 0><<<dim3((unsigned)(M / 16), (unsigned)((Co >> 4) / 2)),
                               dim3(32), 0, stream>>>(
          in, WB, out, N, H, W, Ci, Ho, Wo, Co, KH, KW, st, pad);
    }
  };
  auto bn = [&](float* x, const float* g, const float* b, const float* res,
                int relu, long M, int C) {
    stats_k<<<dim3((unsigned)C), dim3(256), 0, stream>>>(x, MV, M, C);
    long tot = M * (long)C;
    bn_apply_k<<<dim3((unsigned)((tot + 255) / 256)), dim3(256), 0, stream>>>(
        x, MV, g, b, res, relu, M, C);
  };
  // basic_block: base -> w1,w2,g1,b1,g2,b2[,wd,gd,bd]
  auto block = [&](int base, const float* x, float* t1, float* t2, float* td,
                   int N, int H, int Ci, int Co, int st) -> float* {
    int Ho = (H + 2 - 3) / st + 1;
    long M = (long)N * Ho * Ho;
    cvtw(F(base + 0), Co, Ci, Ci, 3, 3);
    conv(x, t1, N, H, H, Ci, Co, 3, 3, st, 1, 0);
    bn(t1, F(base + 2), F(base + 3), nullptr, 1, M, Co);
    cvtw(F(base + 1), Co, Co, Co, 3, 3);
    conv(t1, t2, N, Ho, Ho, Co, Co, 3, 3, 1, 1, 0);
    const float* ident = x;
    if (st != 1 || Ci != Co) {
      cvtw(F(base + 6), Co, Ci, Ci, 1, 1);
      conv(x, td, N, H, H, Ci, Co, 1, 1, st, 0, 0);
      bn(td, F(base + 7), F(base + 8), nullptr, 0, M, Co);
      ident = td;
    }
    bn(t2, F(base + 4), F(base + 5), ident, 1, M, Co);
    return t2;
  };

  // ---- backbone ----
  {
    long tot = 8L * 224 * 224 * 32;
    nchw_to_nhwc_pad_k<<<dim3((unsigned)((tot + 255) / 256)), dim3(256), 0, stream>>>(
        F(0), XN, 8, 3, 32, 224, 224);
  }
  cvtw(F(1), 64, 3, 32, 7, 7);                 // conv1 weights, channels padded 3->32
  conv(XN, A0, 8, 224, 224, 32, 64, 7, 7, 2, 3, 0);
  bn(A0, F(2), F(3), nullptr, 1, 8L * 112 * 112, 64);
  {
    long tot = 8L * 56 * 56 * 64;
    maxpool_k<<<dim3((unsigned)((tot + 255) / 256)), dim3(256), 0, stream>>>(
        A0, L0, 8, 112, 112, 64, 56, 56);
  }
  float* x1 = block(4,  L0, L1, L2, L3, 8, 56, 64, 64, 1);    // -> L2
  float* x2 = block(10, x1, L1, L0, L3, 8, 56, 64, 64, 1);    // -> L0
  float* x3 = block(16, x2, L1, L2, L3, 8, 56, 64, 128, 2);   // -> L2 (28x28x128)
  float* x4 = block(25, x3, L1, L0, L3, 8, 28, 128, 128, 1);  // -> L0
  float* x5 = block(31, x4, L1, L2, L3, 8, 28, 128, 256, 2);  // -> L2 (14x14x256)
  float* cm = block(40, x5, L1, L0, L3, 8, 14, 256, 256, 1);  // -> L0 = cmap

  // ---- RPN + boxes ----
  cvtw(F(46), 32, 256, 256, 3, 3);
  conv(cm, R0, 8, 14, 14, 256, 32, 3, 3, 1, 1, 0);
  {
    long tot = 1568L * 32;
    bias_act_k<<<dim3((unsigned)((tot + 255) / 256)), dim3(256), 0, stream>>>(
        R0, F(47), 1, 1568L, 32);
  }
  {
    long tot = 1568L * 2;
    conv1x1_small_k<<<dim3((unsigned)((tot + 255) / 256)), dim3(256), 0, stream>>>(
        R0, F(48), F(49), o_scores, 1568L, 32, 2);
  }
  {
    long tot = 1568L * 4;
    conv1x1_small_k<<<dim3((unsigned)((tot + 255) / 256)), dim3(256), 0, stream>>>(
        R0, F(50), F(51), o_deltas, 1568L, 32, 4);
  }
  boxes_k<<<dim3((8 * 196 + 255) / 256), dim3(256), 0, stream>>>(o_deltas, BOX, NB, 8, 196);

  // ---- crop & resize ----
  {
    long tot = 1568L * 49 * 256;
    crop_k<<<dim3((unsigned)((tot + 255) / 256)), dim3(256), 0, stream>>>(
        cm, NB, CROPS, 8, 196, 256, 14, 7);
  }

  // ---- mask head ----  (pre-BN channel biases cancel under batch-norm; skipped)
  cvtw(F(52), 256, 256, 256, 3, 3);
  conv(CROPS, MH1, 1568, 7, 7, 256, 256, 3, 3, 1, 1, 0);
  bn(MH1, F(54), F(55), nullptr, 1, 76832L, 256);
  cvtw(F(56), 128, 256, 256, 3, 3);
  conv(MH1, MH2, 1568, 7, 7, 256, 128, 3, 3, 1, 1, 0);
  bn(MH2, F(58), F(59), nullptr, 1, 76832L, 128);
  {
    long tot = 128L * 512;
    wpack_deconv_k<<<dim3((unsigned)((tot + 255) / 256)), dim3(256), 0, stream>>>(
        F(60), WB, 128, 128);
  }
  conv(MH2, UP, 1568, 7, 7, 128, 512, 1, 1, 1, 0, /*store_mode=*/1);  // -> [1568,14,14,128]
  bn(UP, F(62), F(63), nullptr, 1, 1568L * 196, 128);
  {
    long tot = 1568L * 2 * 196;
    mask_final_k<<<dim3((unsigned)((tot + 255) / 256)), dim3(256), 0, stream>>>(
        UP, F(64), F(65), o_masks, 1568, 128, 2);
  }

  // ---- NMS ----
  nms_k<<<dim3(8), dim3(256), 0, stream>>>(o_scores, BOX, o_keep, 196);
}
